// NaiveSSMLayer_73564199846502
// MI455X (gfx1250) — matrix operand order repaired and checked by reading, not compiled, not run on hardware
//
#include <hip/hip_runtime.h>
#include <hip/hip_bf16.h>

// MI455X (gfx1250) SSM layer:
//   prep   : Abar = inv(I-hA)(I+hA), BL, S = (Abar^64)^T        (1 block)
//   kgen_u : U[d,i,:] = Abar^i Bbar[d]      (depth-64 recurrence, f32)
//   kgen_z : Z[d,j,:] = S^j C[d]            (depth-64 recurrence, f32)
//   kmm    : K[d] = Z[d] @ U[d]^T           (WMMA bf16) -> stored REVERSED+padded
//   conv   : block-Toeplitz causal conv     (WMMA bf16, f32 accum, TDM-staged K)
//
// Roofline: conv dominates at ~3.4e10 MACs (69 GFLOP) -> trivially WMMA-bound;
// memory floor is u(64MB)+out(64MB) ~ 5.5us @ 23.3 TB/s; u re-reads across the
// 64 chunk-blocks of each d stay resident in the 192MB L2.

typedef __attribute__((ext_vector_type(16))) __bf16 v16bf;
typedef __attribute__((ext_vector_type(8)))  float  v8f;
typedef __attribute__((ext_vector_type(4)))  unsigned int u32x4;
typedef __attribute__((ext_vector_type(8)))  int    i32x8;
typedef __attribute__((ext_vector_type(4)))  int    i32x4;

#define BATCH  8
#define DMODEL 512
#define LSEQ   4096
#define NST    64
#define RLEN   4352   // reversed-K row stride: 4096 + 256 zero-pad (16B multiple)

// 16-bit A-matrix 16x32 per-lane K index (ISA 7.12.2):
// lanes 0-15: VGPR0-3 -> K=0..7, VGPR4-7 -> K=16..23 ; lanes 16-31: +8
__device__ __forceinline__ int a_k_of(int lane, int e) {
  return ((lane >> 4) << 3) + (e & 7) + ((e >> 3) << 4);
}
// Contiguous per-lane K index (unused here; not AMD's table order): lanes 0-15 K=0..15, lanes 16-31 K=16..31
__device__ __forceinline__ int b_k_of(int lane, int e) {
  return ((lane >> 4) << 4) + e;
}

// ---------------------------------------------------------------- prep
__global__ void ssm_prep(const float* __restrict__ A,
                         float* __restrict__ AbarG,
                         float* __restrict__ BLG,
                         float* __restrict__ SG) {
  __shared__ float M1[64][65];
  __shared__ float W1[64][65];
  __shared__ float T1[64][65];
  const int t = threadIdx.x;  // 0..63, thread t owns row t
  const float step = 1.0f / (float)LSEQ;
  const float h = 0.5f * step;

  for (int j = 0; j < 64; ++j) {
    float a = A[t * 64 + j];
    M1[t][j] = ((t == j) ? 1.0f : 0.0f) - h * a;
    W1[t][j] = (t == j) ? 1.0f : 0.0f;
  }
  __syncthreads();

  // Gauss-Jordan (M ~= I, diagonally dominant; no pivoting needed)
  for (int k = 0; k < 64; ++k) {
    if (t == k) {
      float inv = 1.0f / M1[k][k];
      for (int j = 0; j < 64; ++j) { M1[k][j] *= inv; W1[k][j] *= inv; }
    }
    __syncthreads();
    if (t != k) {
      float f = M1[t][k];
      for (int j = 0; j < 64; ++j) { M1[t][j] -= f * M1[k][j]; W1[t][j] -= f * W1[k][j]; }
    }
    __syncthreads();
  }

  for (int j = 0; j < 64; ++j) BLG[t * 64 + j] = W1[t][j];

  // Abar = BL @ (I + hA)
  for (int j = 0; j < 64; ++j) {
    float s = 0.0f;
    for (int m = 0; m < 64; ++m) s += W1[t][m] * A[m * 64 + j];
    T1[t][j] = W1[t][j] + h * s;
  }
  __syncthreads();
  for (int j = 0; j < 64; ++j) AbarG[t * 64 + j] = T1[t][j];

  // Abar^64 via 6 squarings (tmp in M1)
  for (int it = 0; it < 6; ++it) {
    for (int j = 0; j < 64; ++j) {
      float s = 0.0f;
      for (int m = 0; m < 64; ++m) s += T1[t][m] * T1[m][j];
      M1[t][j] = s;
    }
    __syncthreads();
    for (int j = 0; j < 64; ++j) T1[t][j] = M1[t][j];
    __syncthreads();
  }
  // S = (Abar^64)^T
  for (int j = 0; j < 64; ++j) SG[j * 64 + t] = T1[t][j];
}

// ---------------------------------------------------------------- kgen_u
__global__ void ssm_kgen_u(const float* __restrict__ Bg,
                           const float* __restrict__ AbarG,
                           const float* __restrict__ BLG,
                           __bf16* __restrict__ Ubf) {
  __shared__ float Ab[64 * 65];
  __shared__ float BLs[64 * 65];
  __shared__ float bv[64];
  __shared__ float v[64];
  const int t = threadIdx.x;      // state index n
  const int d = blockIdx.x;       // channel
  for (int q = 0; q < 64; ++q) {
    Ab[q * 65 + t]  = AbarG[q * 64 + t];
    BLs[q * 65 + t] = BLG[q * 64 + t];
  }
  bv[t] = Bg[d * 64 + t];
  __syncthreads();
  const float step = 1.0f / (float)LSEQ;
  float s = 0.0f;
  for (int m = 0; m < 64; ++m) s += BLs[t * 65 + m] * bv[m];
  v[t] = step * s;                // v0 = Bbar[d]
  __syncthreads();
  for (int i = 0; i < 64; ++i) {
    Ubf[(size_t)d * 4096 + i * 64 + t] = (__bf16)v[t];
    float nv = 0.0f;
    for (int m = 0; m < 64; ++m) nv += Ab[t * 65 + m] * v[m];
    __syncthreads();
    v[t] = nv;
    __syncthreads();
  }
}

// ---------------------------------------------------------------- kgen_z
__global__ void ssm_kgen_z(const float* __restrict__ Cg,
                           const float* __restrict__ SG,
                           __bf16* __restrict__ Zbf) {
  __shared__ float Ss[64 * 65];
  __shared__ float v[64];
  const int t = threadIdx.x;
  const int d = blockIdx.x;
  for (int q = 0; q < 64; ++q) Ss[q * 65 + t] = SG[q * 64 + t];
  v[t] = Cg[d * 64 + t];          // z0 = C[d]
  __syncthreads();
  for (int j = 0; j < 64; ++j) {
    Zbf[(size_t)d * 4096 + j * 64 + t] = (__bf16)v[t];
    float nv = 0.0f;
    for (int m = 0; m < 64; ++m) nv += Ss[t * 65 + m] * v[m];
    __syncthreads();
    v[t] = nv;
    __syncthreads();
  }
}

// ---------------------------------------------------------------- kmm
// K[d][j,i] = sum_n Z[d,j,n]*U[d,i,n]  (= K[d, l], l = 64j+i)
// Stored REVERSED with zero pad:  KbfR[d][x] = K[d, 4095-x], x<4096; 0 else.
__global__ void __launch_bounds__(128) ssm_kmm(const __bf16* __restrict__ Zbf,
                                               const __bf16* __restrict__ Ubf,
                                               __bf16* __restrict__ KbfR) {
  __shared__ __bf16 Zl[4096];
  __shared__ __bf16 Ul[4096];
  const int tid = threadIdx.x;    // 0..127 (4 waves)
  const int d = blockIdx.x;
  for (int q = 0; q < 32; ++q) {
    Zl[tid + 128 * q] = Zbf[(size_t)d * 4096 + tid + 128 * q];
    Ul[tid + 128 * q] = Ubf[(size_t)d * 4096 + tid + 128 * q];
  }
  __syncthreads();
  const int w = tid >> 5;         // j-tile
  const int lane = tid & 31;
  const int mrow = lane & 15;
  for (int ti = 0; ti < 4; ++ti) { // i-tile
    v8f acc = {};
#pragma unroll
    for (int kt = 0; kt < 2; ++kt) {
      v16bf a, b;
#pragma unroll
      for (int e = 0; e < 16; ++e) {
        a[e] = Zl[(16 * w + mrow) * 64 + 32 * kt + a_k_of(lane, e)];
        b[e] = Ul[(16 * ti + mrow) * 64 + 32 * kt + a_k_of(lane, e)];
      }
      acc = __builtin_amdgcn_wmma_f32_16x16x32_bf16(false, a, false, b,
                                                    (short)0, acc, false, false);
    }
    const int mhi = (lane >> 4) << 3;
#pragma unroll
    for (int r = 0; r < 8; ++r) {
      const int l = (16 * w + mhi + r) * 64 + (16 * ti + mrow);
      KbfR[(size_t)d * RLEN + (4095 - l)] = (__bf16)acc[r];
    }
  }
  // zero pad so Toeplitz rows with negative lag read 0
  for (int x = 4096 + tid; x < RLEN; x += 128)
    KbfR[(size_t)d * RLEN + x] = (__bf16)0.0f;
}

// ---------------------------------------------------------------- conv
// out[b,d,64c+i] = sum_{g=0..c} sum_p K[d,64g+i-p]*u[b,d,64(c-g)+p] + D[d]*u
__global__ void __launch_bounds__(128) ssm_conv(const float* __restrict__ u,
                                                const __bf16* __restrict__ KbfR,
                                                const float* __restrict__ Dg,
                                                float* __restrict__ out) {
  __shared__ __attribute__((aligned(16))) __bf16 Krev[RLEN];     // reversed + pad
  __shared__ __attribute__((aligned(16))) __bf16 Ub[16 * 64];    // [b][p]
  const int tid = threadIdx.x;    // 0..127
  const int c = blockIdx.x;       // output time chunk
  const int d = blockIdx.y;       // channel

  // ---- stage reversed-K row via Tensor Data Mover (one 8.7KB 1D tile) ----
  if (tid < 32) {                 // wave 0 only; TDM issues once per wave
    const unsigned long long ga =
        (unsigned long long)(uintptr_t)(KbfR + (size_t)d * RLEN);
    const unsigned int lds = (unsigned int)(uintptr_t)(&Krev[0]);
    const unsigned int ndw = (RLEN * 2) / 4;      // 2176 dwords
    u32x4 g0;
    g0[0] = 1u;                                   // count=1 (valid descriptor)
    g0[1] = lds;                                  // lds_addr
    g0[2] = (unsigned int)(ga & 0xffffffffull);   // global_addr[31:0]
    g0[3] = ((unsigned int)(ga >> 32) & 0x01ffffffu) | 0x80000000u; // addr hi | type=2
    i32x8 g1;
    g1[0] = (int)(2u << 16);                      // data_size=4B, no mask/pad/iter
    g1[1] = (int)((ndw & 0xffffu) << 16);         // tensor_dim0[15:0]
    g1[2] = (int)((ndw >> 16) | (1u << 16));      // tensor_dim0[31:16] | tensor_dim1=1
    g1[3] = (int)(ndw << 16);                     // tile_dim0 = ndw
    g1[4] = 0;                                    // tile_dim1=0, tile_dim2=0
    g1[5] = (int)ndw;                             // tensor_dim0_stride[31:0]
    g1[6] = 0;
    g1[7] = 0;
    i32x4 gz4 = {0, 0, 0, 0};
    i32x8 gz8 = {0, 0, 0, 0, 0, 0, 0, 0};
    __builtin_amdgcn_tensor_load_to_lds(g0, g1, gz4, gz4, gz8, 0);
    __builtin_amdgcn_s_wait_tensorcnt(0);
  }
  const int w = tid >> 5;
  const int lane = tid & 31;
  const int mrow = lane & 15;
  const int i0 = 16 * w;          // wave's output rows [i0, i0+16)
  const int idx4 = tid * 4;       // staging: 512 floats, 4 per thread
  const int bb = idx4 >> 6;
  const int p0 = idx4 & 63;
  const size_t urow = ((size_t)(bb * DMODEL + d)) * LSEQ;
  v8f acc = {};
  __syncthreads();

  for (int g = c; g >= 0; --g) {
    const int chunk = c - g;      // input chunk, ascending 0..c
    const float4 uq = *reinterpret_cast<const float4*>(u + urow + chunk * 64 + p0);
    union { __bf16 h[4]; uint2 v2; } pk;
    pk.h[0] = (__bf16)uq.x; pk.h[1] = (__bf16)uq.y;
    pk.h[2] = (__bf16)uq.z; pk.h[3] = (__bf16)uq.w;
    *reinterpret_cast<uint2*>(&Ub[bb * 64 + p0]) = pk.v2;   // [b][p], 8B store
    if (chunk < c)                // pull next chunk toward L2/L0
      __builtin_prefetch(u + urow + (chunk + 1) * 64 + p0, 0, 1);
    __syncthreads();
#pragma unroll
    for (int kt = 0; kt < 2; ++kt) {
      // Toeplitz A from reversed K: branch-free, contiguous-in-e gather
      const int krbase = 4095 - 64 * g - i0 - mrow + 32 * kt + ((lane >> 4) << 3);
      v16bf a, b;
#pragma unroll
      for (int e = 0; e < 16; ++e) {
        a[e] = Krev[krbase + (e & 7) + ((e >> 3) << 4)];
        b[e] = Ub[mrow * 64 + 32 * kt + a_k_of(lane, e)];   // same k order as A
      }
      acc = __builtin_amdgcn_wmma_f32_16x16x32_bf16(false, a, false, b,
                                                    (short)0, acc, false, false);
    }
    __syncthreads();
  }

  // epilogue: N-dim lanes 0..7 are valid batches; add D*u in f32
  if (mrow < BATCH) {
    const float Dd = Dg[d];
    const int mhi = (lane >> 4) << 3;
#pragma unroll
    for (int r = 0; r < 8; ++r) {
      const int i = i0 + mhi + r;
      const size_t idx = ((size_t)(mrow * DMODEL + d)) * LSEQ + (size_t)(c * 64 + i);
      out[idx] = acc[r] + Dd * u[idx];
    }
  }
}

// ---------------------------------------------------------------- launch
extern "C" void kernel_launch(void* const* d_in, const int* in_sizes, int n_in,
                              void* d_out, int out_size, void* d_ws, size_t ws_size,
                              hipStream_t stream) {
  (void)in_sizes; (void)n_in; (void)out_size; (void)ws_size;
  const float* u = (const float*)d_in[0];
  const float* A = (const float*)d_in[1];
  const float* B = (const float*)d_in[2];
  const float* C = (const float*)d_in[3];
  const float* D = (const float*)d_in[4];
  float* out = (float*)d_out;

  char* wsb = (char*)d_ws;
  float*  Abar = (float*)(wsb);                        // 16 KB
  float*  BLw  = (float*)(wsb + 16384);                // 16 KB
  float*  Sw   = (float*)(wsb + 32768);                // 16 KB
  __bf16* Ubf  = (__bf16*)(wsb + 65536);               // 4 MB
  __bf16* Zbf  = (__bf16*)(wsb + 65536 + (1u << 22));  // 4 MB
  __bf16* KbfR = (__bf16*)(wsb + 65536 + (2u << 22));  // 512*4352*2 = 4.46 MB

  ssm_prep  <<<1,   64, 0, stream>>>(A, Abar, BLw, Sw);
  ssm_kgen_u<<<512, 64, 0, stream>>>(B, Abar, BLw, Ubf);
  ssm_kgen_z<<<512, 64, 0, stream>>>(C, Sw, Zbf);
  ssm_kmm   <<<512, 128, 0, stream>>>(Zbf, Ubf, KbfR);
  ssm_conv  <<<dim3(64, 512), 128, 0, stream>>>(u, KbfR, D, out);
}
